// SkelPointNet_52785148068540
// MI455X (gfx1250) — matrix operand order repaired and checked by reading, not compiled, not run on hardware
//
#include <hip/hip_runtime.h>
#include <hip/hip_bf16.h>
#include <stdint.h>

typedef _Float16 f16;
typedef __attribute__((ext_vector_type(8)))  _Float16 v8h;
typedef __attribute__((ext_vector_type(16))) _Float16 v16h;
typedef __attribute__((ext_vector_type(8)))  float    v8f;

#define BN_EPS 1e-5f
static __host__ __device__ inline int align32i(int x) { return (x + 31) & ~31; }

// ---------------------------------------------------------------------------
// Farthest point sampling: one block per batch, iterative argmax.
// mind[] lives in global scratch (32KB per batch -> L2 resident).
// ---------------------------------------------------------------------------
__global__ __launch_bounds__(256) void fps_kernel(const float* __restrict__ xyz,
                                                  float* __restrict__ mind,
                                                  int* __restrict__ ctr,
                                                  int N, int npoint)
{
    const int b = blockIdx.x;
    const int tid = threadIdx.x;
    const float* px = xyz + (size_t)b * N * 3;
    float* md = mind + (size_t)b * N;
    int* ci = ctr + (size_t)b * npoint;

    __shared__ float sv[256];
    __shared__ int   si[256];
    __shared__ int   s_last;

    for (int i = tid; i < N; i += 256) md[i] = 1e10f;
    if (tid == 0) { ci[0] = 0; s_last = 0; }
    __syncthreads();

    for (int it = 1; it < npoint; ++it) {
        const int last = s_last;
        const float lx = px[last * 3 + 0];
        const float ly = px[last * 3 + 1];
        const float lz = px[last * 3 + 2];
        float best = -1.0f;
        int bidx = 0;
        for (int i = tid; i < N; i += 256) {
            float dx = px[i * 3 + 0] - lx;
            float dy = px[i * 3 + 1] - ly;
            float dz = px[i * 3 + 2] - lz;
            float d = dx * dx + dy * dy + dz * dz;
            float m = fminf(md[i], d);
            md[i] = m;
            if (m > best) { best = m; bidx = i; }
        }
        sv[tid] = best; si[tid] = bidx;
        __syncthreads();
        for (int off = 128; off > 0; off >>= 1) {
            if (tid < off && sv[tid + off] > sv[tid]) {
                sv[tid] = sv[tid + off];
                si[tid] = si[tid + off];
            }
            __syncthreads();
        }
        if (tid == 0) { ci[it] = si[0]; s_last = si[0]; }
        __syncthreads();
    }
}

// Gather center coordinates: out[b,s,:] = xyz[b, ctr[b,s], :]
__global__ void gather_xyz_kernel(const float* __restrict__ xyz,
                                  const int* __restrict__ ctr,
                                  float* __restrict__ out,
                                  int total, int np, int nprev)
{
    int t = blockIdx.x * blockDim.x + threadIdx.x;
    if (t >= total) return;
    int b = t / np;
    int p = ctr[t];
    const float* s = xyz + ((size_t)b * nprev + p) * 3;
    out[(size_t)t * 3 + 0] = s[0];
    out[(size_t)t * 3 + 1] = s[1];
    out[(size_t)t * 3 + 2] = s[2];
}

// Ball query: linear index scan gives the smallest-index neighbors
// (matches jnp.sort of masked indices); pad with first hit (0 if none).
__global__ void ball_query_kernel(const float* __restrict__ xyz,
                                  const float* __restrict__ ctr_xyz,
                                  int* __restrict__ out,
                                  int Bv, int N, int np, int ns, float r2)
{
    int t = blockIdx.x * blockDim.x + threadIdx.x;
    if (t >= Bv * np) return;
    int b = t / np;
    const float cx = ctr_xyz[(size_t)t * 3 + 0];
    const float cy = ctr_xyz[(size_t)t * 3 + 1];
    const float cz = ctr_xyz[(size_t)t * 3 + 2];
    const float* px = xyz + (size_t)b * N * 3;
    int cnt = 0, first = 0;
    for (int n = 0; n < N && cnt < ns; ++n) {
        float dx = px[n * 3 + 0] - cx;
        float dy = px[n * 3 + 1] - cy;
        float dz = px[n * 3 + 2] - cz;
        if (dx * dx + dy * dy + dz * dz < r2) {
            if (cnt == 0) first = n;
            out[(size_t)t * ns + cnt] = n;
            ++cnt;
        }
    }
    int pad = (cnt > 0) ? first : 0;
    for (int j = cnt; j < ns; ++j) out[(size_t)t * ns + j] = pad;
}

// ---------------------------------------------------------------------------
// Materialize gathered + centered group features as padded f16 rows:
//   row r = ((b*np + s)*ns + j):  [xyz_rel(3) | pfeat[gi] (cprev) | 0-pad]
// Coalesced: consecutive threads touch consecutive channels.
// ---------------------------------------------------------------------------
__global__ void gather_group_kernel(const int* __restrict__ gi,
                                    const float* __restrict__ xyz,
                                    const float* __restrict__ ctr_xyz,
                                    const float* __restrict__ pfeat,
                                    f16* __restrict__ out,
                                    int total /*rows*kp*/, int kp, int cprev,
                                    int nprev, int npoint, int ns)
{
    int idx = blockIdx.x * blockDim.x + threadIdx.x;
    if (idx >= total) return;
    int r = idx / kp;
    int k = idx - r * kp;
    int j = r % ns;
    int s = (r / ns) % npoint;
    int b = r / (ns * npoint);
    int p = gi[((size_t)b * npoint + s) * ns + j];
    float v = 0.0f;
    if (k < 3)
        v = xyz[((size_t)b * nprev + p) * 3 + k] - ctr_xyz[((size_t)b * npoint + s) * 3 + k];
    else if (k - 3 < cprev)
        v = pfeat[((size_t)b * nprev + p) * cprev + (k - 3)];
    out[idx] = (f16)v;
}

__global__ void f32_to_f16_kernel(const float* __restrict__ in, f16* __restrict__ out, int n)
{
    int idx = blockIdx.x * blockDim.x + threadIdx.x;
    if (idx < n) out[idx] = (f16)in[idx];
}

// ---------------------------------------------------------------------------
// Dense Linear + BN + ReLU with v_wmma_f32_16x16x32_f16.
// A: f16, row stride kp (multiple of 32, zero/unused-padded tail).
// Block = 256 threads = 8 waves. Wave owns 16 rows x 64 cols (4 subtiles,
// 4 accumulators); block tile = 128 x 64. W col-block staged to LDS as f16.
// Inner loop: 2x global_load_b128 (A) + 8x ds_load_b128 (B) + 4x wmma.
// ---------------------------------------------------------------------------
__global__ __launch_bounds__(256) void wmma_linear_kernel(
    const f16* __restrict__ A,
    const float* __restrict__ W, const float* __restrict__ bias,
    const float* __restrict__ gamma, const float* __restrict__ beta,
    const float* __restrict__ mu, const float* __restrict__ var,
    f16* __restrict__ Yh, float* __restrict__ Yf,
    int rows, int cin, int kp, int cout, int strideOut, int relu)
{
    __shared__ alignas(32) f16 wlds[64 * 512];   // [n * kp + k], 64-column block

    const int tid  = threadIdx.x;
    const int col0 = blockIdx.y * 64;

    // Stage weight block (f32 -> f16); zero-pad K tail rows and OOB columns.
    const int wtot = kp << 6;
    for (int idx = tid; idx < wtot; idx += 256) {
        int n = idx / kp;
        int k = idx - n * kp;
        int c = col0 + n;
        float v = (k < cin && c < cout) ? W[(size_t)k * cout + c] : 0.0f;
        wlds[idx] = (f16)v;
    }
    __syncthreads();

    const int lane    = tid & 31;
    const int wave    = tid >> 5;
    const int nlane   = lane & 15;
    const int halfSel = lane >> 4;
    const int row0    = (blockIdx.x * 8 + wave) * 16;
    const int mrow    = row0 + nlane;
    const int rr      = (mrow < rows) ? mrow : 0;

    const f16* ap = A + (size_t)rr * kp;
    __builtin_prefetch(ap, 0, 1);

    v8f acc[4];
    acc[0] = (v8f){}; acc[1] = (v8f){}; acc[2] = (v8f){}; acc[3] = (v8f){};

    for (int kk = 0; kk < kp; kk += 32) {
        // A fragment: lanes 0-15 K kk..kk+15,
        //             lanes 16-31 K kk+16..kk+31.
        v8h alo = *(const v8h*)(ap + kk + halfSel * 16);
        v8h ahi = *(const v8h*)(ap + kk + halfSel * 16 + 8);
        v16h a = __builtin_shufflevector(alo, ahi,
                                         0, 1, 2, 3, 4, 5, 6, 7,
                                         8, 9, 10, 11, 12, 13, 14, 15);
        const f16* bp = wlds + (size_t)nlane * kp + kk + halfSel * 16;
#pragma unroll
        for (int c4 = 0; c4 < 4; ++c4) {
            // B fragment: lane = column; lanes 0-15 K kk..kk+15, 16-31 K kk+16..kk+31.
            v16h b = *(const v16h*)(bp + (size_t)(c4 * 16) * kp);
            acc[c4] = __builtin_amdgcn_wmma_f32_16x16x32_f16(false, a, false, b,
                                                             (short)0, acc[c4], false, false);
        }
    }

    // Epilogue: fold bias + batchnorm into per-column scale/shift, optional ReLU.
    const int mbase = row0 + halfSel * 8;
#pragma unroll
    for (int c4 = 0; c4 < 4; ++c4) {
        const int n = col0 + c4 * 16 + nlane;
        float scale = 0.0f, shift = 0.0f;
        if (n < cout) {
            float rs = rsqrtf(var[n] + BN_EPS);
            scale = gamma[n] * rs;
            shift = (bias[n] - mu[n]) * scale + beta[n];
        }
#pragma unroll
        for (int i = 0; i < 8; ++i) {
            const int m = mbase + i;
            if (m < rows) {
                if (n < cout) {
                    float y = acc[c4][i] * scale + shift;
                    if (relu) y = fmaxf(y, 0.0f);
                    if (Yh) Yh[(size_t)m * strideOut + n] = (f16)y;
                    if (Yf) Yf[(size_t)m * strideOut + n] = y;
                } else if (Yh && n < strideOut) {
                    Yh[(size_t)m * strideOut + n] = (f16)0.0f;  // keep pad finite
                }
            }
        }
    }
}

// Max-pool over the neighbor axis; writes f32 into concatenated feature buffer.
__global__ void maxpool_kernel(const f16* __restrict__ in, float* __restrict__ out,
                               int npts, int ns, int c, int strideIn,
                               int cStrideOut, int cOff)
{
    int idx = blockIdx.x * blockDim.x + threadIdx.x;
    if (idx >= npts * c) return;
    int s = idx / c;
    int ch = idx - s * c;
    const f16* p = in + (size_t)s * ns * strideIn + ch;
    float m = -3.4e38f;
    for (int j = 0; j < ns; ++j) m = fmaxf(m, (float)p[(size_t)j * strideIn]);
    out[(size_t)s * cStrideOut + cOff + ch] = m;
}

// Softmax over axis=1 (the S=512 points) for each (b, k) column.
__global__ __launch_bounds__(256) void softmax_axis1_kernel(const float* __restrict__ L,
                                                            float* __restrict__ w,
                                                            int S, int Kc)
{
    int b = blockIdx.x / Kc;
    int k = blockIdx.x % Kc;
    int tid = threadIdx.x;
    __shared__ float red[256];
    const float* Lp = L + (size_t)b * S * Kc + k;

    float mx = -3.4e38f;
    for (int s = tid; s < S; s += 256) mx = fmaxf(mx, Lp[(size_t)s * Kc]);
    red[tid] = mx; __syncthreads();
    for (int off = 128; off > 0; off >>= 1) {
        if (tid < off) red[tid] = fmaxf(red[tid], red[tid + off]);
        __syncthreads();
    }
    mx = red[0]; __syncthreads();

    float sum = 0.0f;
    for (int s = tid; s < S; s += 256) sum += __expf(Lp[(size_t)s * Kc] - mx);
    red[tid] = sum; __syncthreads();
    for (int off = 128; off > 0; off >>= 1) {
        if (tid < off) red[tid] += red[tid + off];
        __syncthreads();
    }
    float inv = 1.0f / red[0];
    float* wp = w + (size_t)b * S * Kc + k;
    for (int s = tid; s < S; s += 256)
        wp[(size_t)s * Kc] = __expf(Lp[(size_t)s * Kc] - mx) * inv;
}

// out[b,k,c] = sum_s w[b,s,k] * X[b,s,c]  (block per (b,k); w column in LDS)
__global__ __launch_bounds__(128) void weighted_sum_kernel(const float* __restrict__ w,
                                                           const float* __restrict__ X,
                                                           float* __restrict__ out,
                                                           int S, int Kc, int C)
{
    int b = blockIdx.x / Kc;
    int k = blockIdx.x % Kc;
    int tid = threadIdx.x;
    __shared__ float sw[512];
    for (int s = tid; s < S; s += 128) sw[s] = w[((size_t)b * S + s) * Kc + k];
    __syncthreads();
    for (int c = tid; c < C; c += 128) {
        float acc = 0.0f;
        for (int s = 0; s < S; ++s) acc += sw[s] * X[((size_t)b * S + s) * C + c];
        out[((size_t)b * Kc + k) * C + c] = acc;
    }
}

// min over k of sqrt(||xyz[b,s] - skel[b,k]||^2 + 1e-12)
__global__ void mind_kernel(const float* __restrict__ xyz,
                            const float* __restrict__ skel,
                            float* __restrict__ out,
                            int total, int S, int Kc)
{
    int t = blockIdx.x * blockDim.x + threadIdx.x;
    if (t >= total) return;
    int b = t / S;
    float x = xyz[(size_t)t * 3 + 0];
    float y = xyz[(size_t)t * 3 + 1];
    float z = xyz[(size_t)t * 3 + 2];
    const float* sk = skel + (size_t)b * Kc * 3;
    float m = 3.4e38f;
    for (int k = 0; k < Kc; ++k) {
        float dx = x - sk[k * 3 + 0];
        float dy = y - sk[k * 3 + 1];
        float dz = z - sk[k * 3 + 2];
        m = fminf(m, sqrtf(dx * dx + dy * dy + dz * dz + 1e-12f));
    }
    out[t] = m;
}

// ---------------------------------------------------------------------------
// Host orchestration
// ---------------------------------------------------------------------------
extern "C" void kernel_launch(void* const* d_in, const int* in_sizes, int n_in,
                              void* d_out, int out_size, void* d_ws, size_t ws_size,
                              hipStream_t stream)
{
    (void)in_sizes; (void)n_in; (void)out_size; (void)ws_size;
    const int Bv = 4, Nv = 8192, Kc = 100;
    static const int   npts[4]     = {1024, 768, 512, 512};
    static const float radii[4][2] = {{0.1f, 0.2f}, {0.2f, 0.4f}, {0.4f, 0.6f}, {0.6f, 0.8f}};
    static const int   nsmp[4][2]  = {{16, 32}, {32, 64}, {32, 64}, {64, 128}};
    static const int   dims[4][4]  = {{3, 16, 16, 32}, {67, 32, 32, 64},
                                      {131, 64, 64, 128}, {259, 128, 128, 256}};
    static const int   cvx[6]      = {512, 384, 256, 256, 128, 100};

    size_t off = 0;
    auto alloc = [&](size_t bytes) -> void* {
        void* p = (void*)((char*)d_ws + off);
        off += (bytes + 255) & ~(size_t)255;
        return p;
    };

    float* mind = (float*)alloc((size_t)Bv * Nv * sizeof(float));
    const size_t rows_max = (size_t)Bv * 512 * 128;          // level 4, scale 2
    f16* ping = (f16*)alloc(rows_max * 288 * sizeof(f16));   // 288 = max padded width
    f16* pong = (f16*)alloc(rows_max * 288 * sizeof(f16));

    int pidx = 1;  // d_in[0] = input_pc; then (W,b,g,be,mu,var) per layer in tree order
    auto gemm = [&](const f16* A, f16* Yh, float* Yf,
                    int rows, int cin, int cout, int relu) {
        int kp = align32i(cin);
        int so = Yh ? align32i(cout) : cout;
        dim3 g((rows + 127) / 128, (cout + 63) / 64);
        wmma_linear_kernel<<<g, 256, 0, stream>>>(
            A,
            (const float*)d_in[pidx + 0], (const float*)d_in[pidx + 1],
            (const float*)d_in[pidx + 2], (const float*)d_in[pidx + 3],
            (const float*)d_in[pidx + 4], (const float*)d_in[pidx + 5],
            Yh, Yf, rows, cin, kp, cout, so, relu);
        pidx += 6;
    };

    const float* xyz_cur  = (const float*)d_in[0];   // (B, 8192, 3)
    const float* feat_cur = nullptr;
    int n_cur = Nv, c_cur = 0;

    for (int lev = 0; lev < 4; ++lev) {
        const int np = npts[lev];
        int* ctr = (int*)alloc((size_t)Bv * np * sizeof(int));
        fps_kernel<<<Bv, 256, 0, stream>>>(xyz_cur, mind, ctr, n_cur, np);

        float* nxyz = (float*)alloc((size_t)Bv * np * 3 * sizeof(float));
        {
            int tot = Bv * np;
            gather_xyz_kernel<<<(tot + 255) / 256, 256, 0, stream>>>(
                xyz_cur, ctr, nxyz, tot, np, n_cur);
        }

        const int cFeat = 2 * dims[lev][3];
        float* nfeat = (float*)alloc((size_t)Bv * np * cFeat * sizeof(float));
        int cOff = 0;
        for (int sc = 0; sc < 2; ++sc) {
            const int ns = nsmp[lev][sc];
            const float r = radii[lev][sc];
            int* gi = (int*)alloc((size_t)Bv * np * ns * sizeof(int));
            {
                int tot = Bv * np;
                ball_query_kernel<<<(tot + 255) / 256, 256, 0, stream>>>(
                    xyz_cur, nxyz, gi, Bv, n_cur, np, ns, r * r);
            }
            const int rows = Bv * np * ns;
            const int kp0  = align32i(dims[lev][0]);
            {
                int tot = rows * kp0;
                gather_group_kernel<<<(tot + 255) / 256, 256, 0, stream>>>(
                    gi, xyz_cur, nxyz, feat_cur, ping, tot, kp0, c_cur, n_cur, np, ns);
            }
            gemm(ping, pong, nullptr, rows, dims[lev][0], dims[lev][1], 1);
            gemm(pong, ping, nullptr, rows, dims[lev][1], dims[lev][2], 1);
            gemm(ping, pong, nullptr, rows, dims[lev][2], dims[lev][3], 1);
            {
                int c = dims[lev][3];
                int tot = Bv * np * c;
                maxpool_kernel<<<(tot + 255) / 256, 256, 0, stream>>>(
                    pong, nfeat, Bv * np, ns, c, align32i(c), cFeat, cOff);
            }
            cOff += dims[lev][3];
        }
        xyz_cur = nxyz; feat_cur = nfeat; n_cur = np; c_cur = cFeat;
    }

    // ---- cvx MLP: (B*512) x 512 -> ... -> 100, BN every layer, no ReLU on last
    const int S = n_cur;          // 512
    const int rows = Bv * S;      // 2048
    {
        int tot = rows * cvx[0];  // c_cur == 512, stride already multiple of 32
        f32_to_f16_kernel<<<(tot + 255) / 256, 256, 0, stream>>>(feat_cur, ping, tot);
    }
    float* logits = (float*)alloc((size_t)rows * Kc * sizeof(float));
    gemm(ping, pong, nullptr, rows, cvx[0], cvx[1], 1);
    gemm(pong, ping, nullptr, rows, cvx[1], cvx[2], 1);
    gemm(ping, pong, nullptr, rows, cvx[2], cvx[3], 1);
    gemm(pong, ping, nullptr, rows, cvx[3], cvx[4], 1);
    gemm(ping, nullptr, logits, rows, cvx[4], cvx[5], 0);

    float* w = (float*)alloc((size_t)rows * Kc * sizeof(float));
    softmax_axis1_kernel<<<Bv * Kc, 256, 0, stream>>>(logits, w, S, Kc);

    // Outputs concatenated flat: skel_xyz (B,K,3) | skel_r (B,K,1) | shape_cmb (B,K,512)
    float* out_xyz = (float*)d_out;
    float* out_r   = out_xyz + (size_t)Bv * Kc * 3;
    float* out_cmb = out_r + (size_t)Bv * Kc;

    weighted_sum_kernel<<<Bv * Kc, 128, 0, stream>>>(w, xyz_cur, out_xyz, S, Kc, 3);
    weighted_sum_kernel<<<Bv * Kc, 128, 0, stream>>>(w, feat_cur, out_cmb, S, Kc, c_cur);

    float* mn = (float*)alloc((size_t)Bv * S * sizeof(float));
    mind_kernel<<<(Bv * S + 255) / 256, 256, 0, stream>>>(xyz_cur, out_xyz, mn, Bv * S, S, Kc);
    weighted_sum_kernel<<<Bv * Kc, 128, 0, stream>>>(w, mn, out_r, S, Kc, 1);
}
